// Alignment_15642270892360
// MI455X (gfx1250) — matrix operand order repaired and checked by reading, not compiled, not run on hardware
//
#include <hip/hip_runtime.h>
#include <hip/hip_bf16.h>
#include <math.h>

// Problem constants (match reference)
constexpr int DD  = 300;   // feature dim
constexpr int HH  = 5;     // GAT heads
constexpr int NLY = 2;     // layers
constexpr int NB  = 64;    // batch
constexpr int LL  = 1024;  // text nodes
constexpr int KK  = 64;    // image nodes
constexpr int ET  = 2048;  // text edges
constexpr int EI  = 512;   // image edges
constexpr int CHB = 8;     // text batch chunk for scatter buffer
constexpr int ST  = LL + 1;
constexpr int KPD = 320;   // 300 padded to multiple of 32
constexpr int KPW = 1536;  // 1500 padded to multiple of 32

typedef __attribute__((ext_vector_type(16))) __bf16 v16bf;
typedef __attribute__((ext_vector_type(8)))  __bf16 v8bf;
typedef __attribute__((ext_vector_type(8)))  float  v8f;

__device__ inline __bf16 f2bf(float f) {
  unsigned u = __float_as_uint(f);
  u += 0x7FFFu + ((u >> 16) & 1u);            // round-to-nearest-even
  unsigned short s = (unsigned short)(u >> 16);
  return __builtin_bit_cast(__bf16, s);
}

__device__ inline void atomicMaxFloat(float* addr, float val) {
  if (val >= 0.0f) atomicMax((int*)addr, __float_as_int(val));
  else             atomicMin((unsigned int*)addr, __float_as_uint(val));
}

// ---------------- utility kernels ----------------
__global__ void fill_f32(float* p, long n, float v) {
  long i = (long)blockIdx.x * blockDim.x + threadIdx.x;
  if (i < n) p[i] = v;
}

// copy (nb, rows, DD) -> dst with per-batch row stride S (rows)
__global__ void copy_rows(const float* __restrict__ src, float* __restrict__ dst,
                          int rows, int S, long total) {
  long i = (long)blockIdx.x * blockDim.x + threadIdx.x;
  if (i >= total) return;
  long per = (long)rows * DD;
  long n = i / per, r = i % per;
  dst[n * (long)S * DD + r] = src[i];
}

// f32 (R x Kd) row-major -> bf16 (R x KP) row-major, zero padded
__global__ void cvt_pad(const float* __restrict__ src, __bf16* __restrict__ dst,
                        int Kd, int KP, long total /* R*KP */) {
  long i = (long)blockIdx.x * blockDim.x + threadIdx.x;
  if (i >= total) return;
  long r = i / KP; int c = (int)(i % KP);
  dst[i] = (c < Kd) ? f2bf(src[r * (long)Kd + c])
                    : __builtin_bit_cast(__bf16, (unsigned short)0);
}

// Repack W (D,H,D) -> wpackT[e*KPW + (h*D+d)] = W[d,h,e]/H (bf16, "B-transposed"),
// and fold attention vectors: wsrc[h*D+d] = sum_e W[d,h,e]*att_src[h,e]
__global__ void pack_w(const float* __restrict__ W, const float* __restrict__ asrc,
                       const float* __restrict__ adst, __bf16* __restrict__ wpackT,
                       float* __restrict__ wsrc, float* __restrict__ wdst) {
  int i = blockIdx.x * blockDim.x + threadIdx.x;   // i = h*DD + d
  if (i >= HH * DD) return;
  int h = i / DD, d = i % DD;
  const float invH = 1.0f / (float)HH;
  float s1 = 0.f, s2 = 0.f;
  for (int e = 0; e < DD; ++e) {
    float w = W[((size_t)d * HH + h) * DD + e];
    wpackT[(size_t)e * KPW + i] = f2bf(w * invH);
    s1 += w * asrc[h * DD + e];
    s2 += w * adst[h * DD + e];
  }
  wsrc[i] = s1; wdst[i] = s2;
}

// a_src[(b*rows+l)*H+h] = dot(x[batchBase+b, l, :], wsrc[h,:]) (and a_dst)
__global__ void attn_logits(const float* __restrict__ X, int S, int rows, int batchBase,
                            const float* __restrict__ wsrc, const float* __restrict__ wdst,
                            float* __restrict__ asr, float* __restrict__ adst, long total) {
  long i = (long)blockIdx.x * blockDim.x + threadIdx.x;
  if (i >= total) return;
  int h = (int)(i % HH);
  long bl = i / HH;
  int b = (int)(bl / rows), l = (int)(bl % rows);
  const float* xp = X + ((size_t)(batchBase + b) * S + l) * DD;
  const float* ws = wsrc + h * DD;
  const float* wd = wdst + h * DD;
  float s1 = 0.f, s2 = 0.f;
  for (int d = 0; d < DD; ++d) { float x = xp[d]; s1 += x * ws[d]; s2 += x * wd[d]; }
  asr[i] = s1; adst[i] = s2;
}

// pass 1: e = leaky_relu(a_src[src]+a_dst[dst], 0.2); segment max via atomics
__global__ void edge_max(const int* __restrict__ edges, long ebStride, int E,
                         const float* __restrict__ asr, const float* __restrict__ adst,
                         float* __restrict__ ebuf, float* __restrict__ mbuf,
                         int rows, long total) {
  long i = (long)blockIdx.x * blockDim.x + threadIdx.x;
  if (i >= total) return;
  int h = (int)(i % HH);
  long be = i / HH;
  int b = (int)(be / E), e = (int)(be % E);
  const int* ep = edges + (size_t)b * ebStride;
  int s = ep[e], t = ep[E + e];
  float v = asr[((size_t)b * rows + s) * HH + h] + adst[((size_t)b * rows + t) * HH + h];
  v = v > 0.f ? v : 0.2f * v;
  ebuf[i] = v;
  atomicMaxFloat(&mbuf[((size_t)b * rows + t) * HH + h], v);
}

// pass 2: p = exp(e - m[dst]); denom[dst] += p
__global__ void edge_expsum(const int* __restrict__ edges, long ebStride, int E,
                            float* __restrict__ ebuf, const float* __restrict__ mbuf,
                            float* __restrict__ dbuf, int rows, long total) {
  long i = (long)blockIdx.x * blockDim.x + threadIdx.x;
  if (i >= total) return;
  int h = (int)(i % HH);
  long be = i / HH;
  int b = (int)(be / E), e = (int)(be % E);
  const int* ep = edges + (size_t)b * ebStride;
  int t = ep[E + e];
  float p = expf(ebuf[i] - mbuf[((size_t)b * rows + t) * HH + h]);
  ebuf[i] = p;
  atomicAdd(&dbuf[((size_t)b * rows + t) * HH + h], p);
}

// pass 3: agg[dst,h,:] += alpha_h * x[src,:]   (aggregate in INPUT space)
__global__ void edge_agg(const int* __restrict__ edges, long ebStride, int E,
                         const float* __restrict__ ebuf, const float* __restrict__ dbuf,
                         const float* __restrict__ X, int S, int rows, int batchBase,
                         float* __restrict__ agg) {
  int be = blockIdx.x;
  int b = be / E, e = be % E;
  const int* ep = edges + (size_t)b * ebStride;
  int s = ep[e], t = ep[E + e];
  int d = threadIdx.x;
  if (d >= DD) return;
  float x = X[((size_t)(batchBase + b) * S + s) * DD + d];
  float* ap = agg + ((size_t)b * rows + t) * HH * DD + d;
  const float* pp = ebuf + ((size_t)b * E + e) * HH;
  const float* dn = dbuf + ((size_t)b * rows + t) * HH;
#pragma unroll
  for (int h = 0; h < HH; ++h)
    atomicAdd(ap + h * DD, (pp[h] / (dn[h] + 1e-16f)) * x);
}

// ---------------- WMMA GEMM (pure bf16, padded K, branch-free inner loop) ----
// C(M,Ncol) = alpha * A(M,KP) x B(Ncol,KP)^T ; both row-major bf16, KP % 32 == 0,
// columns [Kd,KP) are zero. One 32x32 macro-tile per wave32 (4 v_wmma per K step).
__global__ void wmma_gemm_bf(const __bf16* __restrict__ Ab, long strideA, int KP, int M,
                             const __bf16* __restrict__ Bb, long strideB, int Ncol,
                             float* __restrict__ Cf, long strideC, int ldc, float alpha) {
  const __bf16* A = Ab + (size_t)blockIdx.z * strideA;
  const __bf16* B = Bb + (size_t)blockIdx.z * strideB;
  float* C = Cf + (size_t)blockIdx.z * strideC;

  int lane = threadIdx.x & 31;
  int idx  = lane & 15, grp = lane >> 4;
  int m0 = blockIdx.y * 32, n0 = blockIdx.x * 32;

  // Clamp load rows/cols: out-of-range rows only feed C entries we never store.
  int ma = m0 + idx;      ma = ma < M ? ma : M - 1;
  int mb = m0 + 16 + idx; mb = mb < M ? mb : M - 1;
  int na = n0 + idx;      na = na < Ncol ? na : Ncol - 1;
  int nb = n0 + 16 + idx; nb = nb < Ncol ? nb : Ncol - 1;
  const __bf16* Ar0 = A + (size_t)ma * KP;
  const __bf16* Ar1 = A + (size_t)mb * KP;
  const __bf16* Br0 = B + (size_t)na * KP;
  const __bf16* Br1 = B + (size_t)nb * KP;

  v8f acc00 = {0,0,0,0,0,0,0,0}, acc01 = {0,0,0,0,0,0,0,0};
  v8f acc10 = {0,0,0,0,0,0,0,0}, acc11 = {0,0,0,0,0,0,0,0};

  for (int k0 = 0; k0 < KP; k0 += 32) {
    if (k0 + 32 < KP) {
      __builtin_prefetch(Ar0 + k0 + 32, 0, 1);   // global_prefetch_b8
      __builtin_prefetch(Br0 + k0 + 32, 0, 1);
    }
    // A fragment: lane needs bf16 chunks [8*grp,8*grp+8) and [16+8*grp,...+8)
    const v8bf* a0p = (const v8bf*)(Ar0 + k0);
    const v8bf* a1p = (const v8bf*)(Ar1 + k0);
    // B fragment: lane needs bf16 chunks [8*grp,8*grp+8) and [16+8*grp,...+8)
    const v8bf* b0p = (const v8bf*)(Br0 + k0);
    const v8bf* b1p = (const v8bf*)(Br1 + k0);
    v8bf a0lo = a0p[grp],     a0hi = a0p[2 + grp];
    v8bf a1lo = a1p[grp],     a1hi = a1p[2 + grp];
    v8bf b0lo = b0p[grp], b0hi = b0p[2 + grp];
    v8bf b1lo = b1p[grp], b1hi = b1p[2 + grp];
    v16bf a0 = __builtin_shufflevector(a0lo, a0hi, 0,1,2,3,4,5,6,7,8,9,10,11,12,13,14,15);
    v16bf a1 = __builtin_shufflevector(a1lo, a1hi, 0,1,2,3,4,5,6,7,8,9,10,11,12,13,14,15);
    v16bf b0 = __builtin_shufflevector(b0lo, b0hi, 0,1,2,3,4,5,6,7,8,9,10,11,12,13,14,15);
    v16bf b1 = __builtin_shufflevector(b1lo, b1hi, 0,1,2,3,4,5,6,7,8,9,10,11,12,13,14,15);
    acc00 = __builtin_amdgcn_wmma_f32_16x16x32_bf16(false, a0, false, b0, (short)0, acc00, false, false);
    acc01 = __builtin_amdgcn_wmma_f32_16x16x32_bf16(false, a0, false, b1, (short)0, acc01, false, false);
    acc10 = __builtin_amdgcn_wmma_f32_16x16x32_bf16(false, a1, false, b0, (short)0, acc10, false, false);
    acc11 = __builtin_amdgcn_wmma_f32_16x16x32_bf16(false, a1, false, b1, (short)0, acc11, false, false);
  }

  // C f32 16x16 tile: VGPR r -> M = r (lanes 0-15) / 8+r (lanes 16-31), lane = N
  auto store_tile = [&](const v8f& acc, int mt, int nt) {
    int n = nt + idx;
    if (n >= Ncol) return;
#pragma unroll
    for (int r = 0; r < 8; ++r) {
      int m = mt + r + 8 * grp;
      if (m < M) C[(size_t)m * ldc + n] = alpha * acc[r];
    }
  };
  store_tile(acc00, m0,      n0);
  store_tile(acc01, m0,      n0 + 16);
  store_tile(acc10, m0 + 16, n0);
  store_tile(acc11, m0 + 16, n0 + 16);
}

// g + bias, gnn_mask zeroing, relu, layernorm -> out (row stride S)
__global__ void post_gat(const float* __restrict__ g, const float* __restrict__ bias,
                         const unsigned char* __restrict__ gmask,
                         const float* __restrict__ lng, const float* __restrict__ lnb,
                         float* __restrict__ out, int rows, int batchBase, int S) {
  int row = blockIdx.x;                 // chunk-local b*rows + l
  int b = row / rows, l = row % rows;
  bool masked = (gmask != nullptr) && (gmask[batchBase + b] != 0);
  const float* gp = g + (size_t)row * DD;
  int tid = threadIdx.x;
  float vloc[3]; float s = 0.f;
#pragma unroll
  for (int i = 0; i < 3; ++i) {
    int d = tid + i * 128; float v = 0.f;
    if (d < DD) { v = masked ? 0.f : gp[d] + bias[d]; v = v > 0.f ? v : 0.f; }
    vloc[i] = v; s += v;
  }
  __shared__ float red[128];
  red[tid] = s; __syncthreads();
  for (int o = 64; o > 0; o >>= 1) { if (tid < o) red[tid] += red[tid + o]; __syncthreads(); }
  float mu = red[0] / (float)DD;
  __syncthreads();
  float s2 = 0.f;
#pragma unroll
  for (int i = 0; i < 3; ++i) { int d = tid + i * 128; if (d < DD) { float t = vloc[i] - mu; s2 += t * t; } }
  red[tid] = s2; __syncthreads();
  for (int o = 64; o > 0; o >>= 1) { if (tid < o) red[tid] += red[tid + o]; __syncthreads(); }
  float inv = rsqrtf(red[0] / (float)DD + 1e-5f);
  float* op = out + ((size_t)(batchBase + b) * S + l) * DD;
#pragma unroll
  for (int i = 0; i < 3; ++i) { int d = tid + i * 128; if (d < DD) op[d] = (vloc[i] - mu) * inv * lng[d] + lnb[d]; }
}

// c[n,d] = sum_l score*t2 -> written into row LL of the final text buffer
__global__ void compute_c(const float* __restrict__ score, const float* __restrict__ t2,
                          float* __restrict__ tfin) {
  int i = blockIdx.x * blockDim.x + threadIdx.x;
  if (i >= NB * DD) return;
  int n = i / DD, d = i % DD;
  float s = 0.f;
  for (int l = 0; l < LL; ++l) {
    size_t o = ((size_t)n * LL + l) * DD + d;
    s += score[o] * t2[o];
  }
  tfin[((size_t)n * ST + LL) * DD + d] = s;
}

// out[n*rows+l] = dot(X[n,l,:], w) + b, masked -> -inf
__global__ void pa_vec(const float* __restrict__ X, int S, int rows,
                       const float* __restrict__ w, const float* __restrict__ bsc,
                       const unsigned char* __restrict__ mask, float* __restrict__ out,
                       long total) {
  long i = (long)blockIdx.x * blockDim.x + threadIdx.x;
  if (i >= total) return;
  int n = (int)(i / rows), l = (int)(i % rows);
  const float* xp = X + ((size_t)n * S + l) * DD;
  float s = bsc[0];
  for (int d = 0; d < DD; ++d) s += xp[d] * w[d];
  out[i] = mask[(size_t)n * rows + l] ? -INFINITY : s;
}

__global__ void softmax_rows(float* __restrict__ p, int rows) {
  int n = blockIdx.x, tid = threadIdx.x;
  float* row = p + (size_t)n * rows;
  __shared__ float red[256];
  float mx = -INFINITY;
  for (int i = tid; i < rows; i += 256) mx = fmaxf(mx, row[i]);
  red[tid] = mx; __syncthreads();
  for (int o = 128; o > 0; o >>= 1) { if (tid < o) red[tid] = fmaxf(red[tid], red[tid + o]); __syncthreads(); }
  mx = red[0]; __syncthreads();
  float s = 0.f;
  for (int i = tid; i < rows; i += 256) { float e = expf(row[i] - mx); row[i] = e; s += e; }
  red[tid] = s; __syncthreads();
  for (int o = 128; o > 0; o >>= 1) { if (tid < o) red[tid] += red[tid + o]; __syncthreads(); }
  float inv = 1.0f / red[0];
  for (int i = tid; i < rows; i += 256) row[i] *= inv;
}

// out[n, outOff+k] = sum_l pa[n,l] * q[n,l,k]   (out row stride 2K)
__global__ void weighted_sum(const float* __restrict__ q, const float* __restrict__ pa,
                             float* __restrict__ out, int rows, int outOff) {
  int n = blockIdx.x, k = threadIdx.x;
  float s = 0.f;
  for (int l = 0; l < rows; ++l)
    s += pa[(size_t)n * rows + l] * q[((size_t)n * rows + l) * KK + k];
  out[(size_t)n * 2 * KK + outOff + k] = s;
}

// ---------------- host orchestration ----------------
extern "C" void kernel_launch(void* const* d_in, const int* in_sizes, int n_in,
                              void* d_out, int out_size, void* d_ws, size_t ws_size,
                              hipStream_t stream) {
  (void)in_sizes; (void)n_in; (void)out_size; (void)ws_size;
  const float* t2    = (const float*)d_in[0];
  const float* v2    = (const float*)d_in[1];
  const float* score = (const float*)d_in[2];
  const int*   eidx  = (const int*)d_in[3];
  const unsigned char* gnn = (const unsigned char*)d_in[4];
  const unsigned char* kpm = (const unsigned char*)d_in[5];
  const unsigned char* npm = (const unsigned char*)d_in[6];
  const int*   imge  = (const int*)d_in[7];
  const float* txtW  = (const float*)d_in[8];
  const float* txtAs = (const float*)d_in[9];
  const float* txtAd = (const float*)d_in[10];
  const float* txtB  = (const float*)d_in[11];
  const float* imgW  = (const float*)d_in[12];
  const float* imgAs = (const float*)d_in[13];
  const float* imgAd = (const float*)d_in[14];
  const float* imgB  = (const float*)d_in[15];
  const float* w1    = (const float*)d_in[16];
  const float* b1    = (const float*)d_in[17];
  const float* w2    = (const float*)d_in[18];
  const float* b2    = (const float*)d_in[19];
  const float* lng   = (const float*)d_in[20];
  const float* lnb   = (const float*)d_in[21];
  float* out = (float*)d_out;

  char* base = (char*)d_ws;
  size_t off = 0;
  auto alloc = [&](size_t bytes) -> void* {
    void* p = base + off;
    off += (bytes + 255) & ~(size_t)255;
    return p;
  };
  float*  tA     = (float*)alloc((size_t)NB * ST * DD * 4);
  float*  tB     = (float*)alloc((size_t)NB * ST * DD * 4);
  float*  vA     = (float*)alloc((size_t)NB * KK * DD * 4);
  float*  vB     = (float*)alloc((size_t)NB * KK * DD * 4);
  float*  agg    = (float*)alloc((size_t)CHB * LL * HH * DD * 4);     // >= NB*KK*HH*DD
  __bf16* aggbf  = (__bf16*)alloc((size_t)CHB * LL * KPW * 2);        // >= NB*KK*KPW
  float*  gbuf   = (float*)alloc((size_t)CHB * LL * DD * 4);          // >= NB*KK*DD
  __bf16* wpackT = (__bf16*)alloc((size_t)DD * KPW * 2);
  __bf16* t2bf   = (__bf16*)alloc((size_t)NB * LL * KPD * 2);
  __bf16* v2bf   = (__bf16*)alloc((size_t)NB * KK * KPD * 2);
  __bf16* tfbf   = (__bf16*)alloc((size_t)NB * ST * KPD * 2);
  __bf16* vfbf   = (__bf16*)alloc((size_t)NB * KK * KPD * 2);
  float*  wsrc   = (float*)alloc((size_t)HH * DD * 4);
  float*  wdst   = (float*)alloc((size_t)HH * DD * 4);
  float*  asr    = (float*)alloc((size_t)CHB * LL * HH * 4);          // >= NB*KK*HH
  float*  adst   = (float*)alloc((size_t)CHB * LL * HH * 4);
  float*  mbuf   = (float*)alloc((size_t)CHB * LL * HH * 4);
  float*  dbuf   = (float*)alloc((size_t)CHB * LL * HH * 4);
  float*  ebuf   = (float*)alloc((size_t)NB * EI * HH * 4);           // >= CHB*ET*HH
  float*  q1     = (float*)alloc((size_t)NB * LL * KK * 4);
  float*  q2     = (float*)alloc((size_t)NB * ST * KK * 4);
  float*  patok  = (float*)alloc((size_t)NB * LL * 4);
  float*  panp   = (float*)alloc((size_t)NB * ST * 4);

  auto fill = [&](float* p, long n, float v) {
    fill_f32<<<dim3((unsigned)((n + 255) / 256)), 256, 0, stream>>>(p, n, v);
  };
  auto cvt = [&](const float* s, __bf16* d, int Kd, int KP, long R) {
    long tot = R * KP;
    cvt_pad<<<dim3((unsigned)((tot + 255) / 256)), 256, 0, stream>>>(s, d, Kd, KP, tot);
  };
  const float scale = 1.0f / sqrtf((float)DD);

  // Stage node features (text buffers have L+1 row stride so 'c' concatenates in place)
  copy_rows<<<(unsigned)(((long)NB * LL * DD + 255) / 256), 256, 0, stream>>>(t2, tA, LL, ST, (long)NB * LL * DD);
  copy_rows<<<(unsigned)(((long)NB * KK * DD + 255) / 256), 256, 0, stream>>>(v2, vA, KK, KK, (long)NB * KK * DD);
  // bf16 staging for q1 operands
  cvt(t2, t2bf, DD, KPD, (long)NB * LL);
  cvt(v2, v2bf, DD, KPD, (long)NB * KK);

  // ---- text GAT (2 layers, chunked over batch) ----
  float* tcur = tA; float* tnxt = tB;
  for (int layer = 0; layer < NLY; ++layer) {
    fill((float*)wpackT, (long)DD * KPW / 2, 0.f);   // zero bf16 pad region
    pack_w<<<(HH * DD + 255) / 256, 256, 0, stream>>>(
        txtW + (size_t)layer * DD * HH * DD, txtAs + (size_t)layer * HH * DD,
        txtAd + (size_t)layer * HH * DD, wpackT, wsrc, wdst);
    for (int c = 0; c < NB / CHB; ++c) {
      int B0 = c * CHB;
      fill(agg, (long)CHB * LL * HH * DD, 0.f);
      fill(mbuf, (long)CHB * LL * HH, -INFINITY);
      fill(dbuf, (long)CHB * LL * HH, 0.f);
      attn_logits<<<(unsigned)(((long)CHB * LL * HH + 255) / 256), 256, 0, stream>>>(
          tcur, ST, LL, B0, wsrc, wdst, asr, adst, (long)CHB * LL * HH);
      const int* ep = eidx + (size_t)B0 * 2 * ET;
      edge_max<<<(unsigned)(((long)CHB * ET * HH + 255) / 256), 256, 0, stream>>>(
          ep, 2 * ET, ET, asr, adst, ebuf, mbuf, LL, (long)CHB * ET * HH);
      edge_expsum<<<(unsigned)(((long)CHB * ET * HH + 255) / 256), 256, 0, stream>>>(
          ep, 2 * ET, ET, ebuf, mbuf, dbuf, LL, (long)CHB * ET * HH);
      edge_agg<<<CHB * ET, 320, 0, stream>>>(ep, 2 * ET, ET, ebuf, dbuf, tcur, ST, LL, B0, agg);
      cvt(agg, aggbf, HH * DD, KPW, (long)CHB * LL);
      wmma_gemm_bf<<<dim3((DD + 31) / 32, LL / 32, CHB), 32, 0, stream>>>(
          aggbf, (long)LL * KPW, KPW, LL, wpackT, 0, DD,
          gbuf, (long)LL * DD, DD, 1.0f);
      post_gat<<<CHB * LL, 128, 0, stream>>>(gbuf, txtB + (size_t)layer * DD, gnn,
                                             lng, lnb, tnxt, LL, B0, ST);
    }
    float* tmp = tcur; tcur = tnxt; tnxt = tmp;
  }
  float* tfin = tcur;

  // ---- image GAT (2 layers, all batches at once, shared edges) ----
  float* vcur = vA; float* vnxt = vB;
  for (int layer = 0; layer < NLY; ++layer) {
    fill((float*)wpackT, (long)DD * KPW / 2, 0.f);
    pack_w<<<(HH * DD + 255) / 256, 256, 0, stream>>>(
        imgW + (size_t)layer * DD * HH * DD, imgAs + (size_t)layer * HH * DD,
        imgAd + (size_t)layer * HH * DD, wpackT, wsrc, wdst);
    fill(agg, (long)NB * KK * HH * DD, 0.f);
    fill(mbuf, (long)NB * KK * HH, -INFINITY);
    fill(dbuf, (long)NB * KK * HH, 0.f);
    attn_logits<<<(unsigned)(((long)NB * KK * HH + 255) / 256), 256, 0, stream>>>(
        vcur, KK, KK, 0, wsrc, wdst, asr, adst, (long)NB * KK * HH);
    edge_max<<<(unsigned)(((long)NB * EI * HH + 255) / 256), 256, 0, stream>>>(
        imge, 0, EI, asr, adst, ebuf, mbuf, KK, (long)NB * EI * HH);
    edge_expsum<<<(unsigned)(((long)NB * EI * HH + 255) / 256), 256, 0, stream>>>(
        imge, 0, EI, ebuf, mbuf, dbuf, KK, (long)NB * EI * HH);
    edge_agg<<<NB * EI, 320, 0, stream>>>(imge, 0, EI, ebuf, dbuf, vcur, KK, KK, 0, agg);
    cvt(agg, aggbf, HH * DD, KPW, (long)NB * KK);
    wmma_gemm_bf<<<dim3((DD + 31) / 32, KK / 32, NB), 32, 0, stream>>>(
        aggbf, (long)KK * KPW, KPW, KK, wpackT, 0, DD,
        gbuf, (long)KK * DD, DD, 1.0f);
    post_gat<<<NB * KK, 128, 0, stream>>>(gbuf, imgB + (size_t)layer * DD, nullptr,
                                          lng, lnb, vnxt, KK, 0, KK);
    float* tmp = vcur; vcur = vnxt; vnxt = tmp;
  }
  float* vfin = vcur;

  // ---- heads: c row, bf16 staging, q1/q2 (WMMA, NT), softmaxes, weighted sums ----
  compute_c<<<(NB * DD + 255) / 256, 256, 0, stream>>>(score, t2, tfin);
  cvt(tfin, tfbf, DD, KPD, (long)NB * ST);
  cvt(vfin, vfbf, DD, KPD, (long)NB * KK);
  wmma_gemm_bf<<<dim3(KK / 32, LL / 32, NB), 32, 0, stream>>>(
      t2bf, (long)LL * KPD, KPD, LL, v2bf, (long)KK * KPD, KK,
      q1, (long)LL * KK, KK, scale);
  wmma_gemm_bf<<<dim3(KK / 32, (ST + 31) / 32, NB), 32, 0, stream>>>(
      tfbf, (long)ST * KPD, KPD, ST, vfbf, (long)KK * KPD, KK,
      q2, (long)ST * KK, KK, scale);
  pa_vec<<<(unsigned)(((long)NB * LL + 255) / 256), 256, 0, stream>>>(
      t2, LL, LL, w1, b1, kpm, patok, (long)NB * LL);
  pa_vec<<<(unsigned)(((long)NB * ST + 255) / 256), 256, 0, stream>>>(
      tfin, ST, ST, w2, b2, npm, panp, (long)NB * ST);
  softmax_rows<<<NB, 256, 0, stream>>>(patok, LL);
  softmax_rows<<<NB, 256, 0, stream>>>(panp, ST);
  weighted_sum<<<NB, KK, 0, stream>>>(q1, patok, out, LL, 0);
  weighted_sum<<<NB, KK, 0, stream>>>(q2, panp, out, ST, KK);
}
